// IDGCL_60361470378156
// MI455X (gfx1250) — compile-verified, matrix operand order checked by reading, not run on hardware
//
#include <hip/hip_runtime.h>
#include <hip/hip_bf16.h>
#include <math.h>

#define D_IN 512
#define H1d  512
#define REP  256
#define PHID 512
#define CHUNK 128
#define EPS_BN 1e-5f
#define NEG_HUGE (-3.402823466e38f)

typedef __attribute__((ext_vector_type(16))) __bf16 v16bf;
typedef __attribute__((ext_vector_type(8)))  float  v8f;

union Frag { v16bf v; uint4 q[2]; };

// ---------------------------------------------------------------- WMMA GEMM
// C[M x Nc] = A[M x K] @ B[K x Nc] (+ bias), with A in bf16 row-major and
// B pre-transposed to bf16 Bt[Nc x K] row-major so both fragments are
// contiguous 16B loads per lane. One wave32 computes a 16x32 C tile
// (one A fragment feeds two WMMAs). Requires M%16==0, K%32==0, Nc%32==0.
__global__ __launch_bounds__(256)
void gemm_bf16_wmma(const __bf16* __restrict__ A, const __bf16* __restrict__ Bt,
                    float* __restrict__ C, const float* __restrict__ bias,
                    int M, int K, int Nc) {
    const int lane    = threadIdx.x & 31;
    const int waveId  = threadIdx.x >> 5;
    const int tilesN2 = Nc >> 5;                  // 32-wide column tiles
    const int tile    = blockIdx.x * 8 + waveId;
    const int tileM   = tile / tilesN2;
    const int tileN   = tile % tilesN2;
    if (tileM * 16 >= M) return;                  // wave-uniform exit

    const int half = lane >> 4;                   // 0 or 1
    const int l15  = lane & 15;
    const __bf16* Arow = A  + (size_t)(tileM * 16 + l15) * K + half * 8;
    const __bf16* B0   = Bt + (size_t)(tileN * 32 + l15) * K + half * 8;
    const __bf16* B1   = B0 + (size_t)16 * K;

    v8f acc0 = {}, acc1 = {};
    for (int k0 = 0; k0 < K; k0 += 32) {
        if (k0 + 32 < K) {
            __builtin_prefetch(Arow + k0 + 32, 0, 1);
            __builtin_prefetch(B0   + k0 + 32, 0, 1);
            __builtin_prefetch(B1   + k0 + 32, 0, 1);
        }
        Frag a, b0, b1;
        a.q[0]  = *(const uint4*)(Arow + k0);
        a.q[1]  = *(const uint4*)(Arow + k0 + 16);
        b0.q[0] = *(const uint4*)(B0 + k0);
        b0.q[1] = *(const uint4*)(B0 + k0 + 16);
        b1.q[0] = *(const uint4*)(B1 + k0);
        b1.q[1] = *(const uint4*)(B1 + k0 + 16);
        acc0 = __builtin_amdgcn_wmma_f32_16x16x32_bf16(
                   false, a.v, false, b0.v, (short)0, acc0, false, false);
        acc1 = __builtin_amdgcn_wmma_f32_16x16x32_bf16(
                   false, a.v, false, b1.v, (short)0, acc1, false, false);
    }

    const int crow0 = tileM * 16 + half * 8;
    const int c0 = tileN * 32 + l15;
    const int c1 = c0 + 16;
    const float bv0 = bias ? bias[c0] : 0.f;
    const float bv1 = bias ? bias[c1] : 0.f;
    #pragma unroll
    for (int v = 0; v < 8; ++v) {
        C[(size_t)(crow0 + v) * Nc + c0] = acc0[v] + bv0;
        C[(size_t)(crow0 + v) * Nc + c1] = acc1[v] + bv1;
    }
}

// ---------------------------------------------------------------- bf16 prep
__global__ void cvt_f32_bf16(const float* __restrict__ x, __bf16* y, size_t n) {
    size_t i = (size_t)blockIdx.x * blockDim.x + threadIdx.x;
    if (i < n) y[i] = (__bf16)x[i];
}

// Bt[n*K + k] = (bf16) B[k*Nc + n]
__global__ void cvt_transpose_bf16(const float* __restrict__ B, __bf16* Bt,
                                   int K, int Nc) {
    int idx = blockIdx.x * blockDim.x + threadIdx.x;
    if (idx >= K * Nc) return;
    int n = idx / K, k = idx % K;
    Bt[idx] = (__bf16)B[(size_t)k * Nc + n];
}

// ---------------------------------------------------------------- GCN pieces
__global__ void deg_init(float* deg, int n) {
    int i = blockIdx.x * blockDim.x + threadIdx.x;
    if (i < n) deg[i] = 1.0f;                    // self loop
}
__global__ void deg_count(const int* __restrict__ col, float* deg, int E) {
    int e = blockIdx.x * blockDim.x + threadIdx.x;
    if (e < E) atomicAdd(&deg[col[e]], 1.0f);
}
__global__ void deg_rsqrt(const float* __restrict__ deg, float* dis, int n) {
    int i = blockIdx.x * blockDim.x + threadIdx.x;
    if (i < n) dis[i] = rsqrtf(fmaxf(deg[i], 1e-12f));
}

// out[c,:] += dis[r]*dis[c] * h[r,:]   (one block per edge)
__global__ __launch_bounds__(128)
void gcn_scatter(const int* __restrict__ row, const int* __restrict__ col,
                 const float* __restrict__ dis, const float* __restrict__ h,
                 float* out, int E, int F) {
    int e = blockIdx.x;
    if (e >= E) return;
    int r = row[e], c = col[e];
    float nrm = dis[r] * dis[c];
    const float* hs = h + (size_t)r * F;
    float* os = out + (size_t)c * F;
    for (int f = threadIdx.x; f < F; f += blockDim.x)
        atomicAdd(&os[f], nrm * hs[f]);
}

// out[i,:] += dis[i]^2 * h[i,:] + b[:]
__global__ void gcn_self_bias(const float* __restrict__ dis,
                              const float* __restrict__ h,
                              const float* __restrict__ b,
                              float* out, int Nn, int F) {
    size_t idx = (size_t)blockIdx.x * blockDim.x + threadIdx.x;
    if (idx >= (size_t)Nn * F) return;
    int i = (int)(idx / F), f = (int)(idx % F);
    out[idx] += dis[i] * dis[i] * h[idx] + b[f];
}

// ---------------------------------------------------------------- BatchNorm
__global__ __launch_bounds__(256)
void bn_stats(const float* __restrict__ x, float* mean, float* rstd,
              int Nn, int F) {
    int f = blockIdx.x;
    float s = 0.f, s2 = 0.f;
    for (int i = threadIdx.x; i < Nn; i += 256) {
        float v = x[(size_t)i * F + f];
        s += v; s2 += v * v;
    }
    __shared__ float sh[256], sh2[256];
    sh[threadIdx.x] = s; sh2[threadIdx.x] = s2;
    __syncthreads();
    for (int o = 128; o > 0; o >>= 1) {
        if (threadIdx.x < o) {
            sh[threadIdx.x]  += sh[threadIdx.x + o];
            sh2[threadIdx.x] += sh2[threadIdx.x + o];
        }
        __syncthreads();
    }
    if (threadIdx.x == 0) {
        float m = sh[0] / (float)Nn;
        float v = sh2[0] / (float)Nn - m * m;
        mean[f] = m;
        rstd[f] = rsqrtf(fmaxf(v, 0.f) + EPS_BN);
    }
}

__global__ void bn_prelu(const float* __restrict__ x,
                         const float* __restrict__ mean,
                         const float* __restrict__ rstd,
                         const float* __restrict__ g,
                         const float* __restrict__ b,
                         const float* __restrict__ alpha,
                         float* y, int Nn, int F) {
    size_t idx = (size_t)blockIdx.x * blockDim.x + threadIdx.x;
    if (idx >= (size_t)Nn * F) return;
    int f = (int)(idx % F);
    float v = (x[idx] - mean[f]) * rstd[f] * g[f] + b[f];
    y[idx] = v > 0.f ? v : alpha[0] * v;
}

// ---------------------------------------------------------------- row kernels
__device__ __forceinline__ float wave_sum(float s) {
    #pragma unroll
    for (int o = 16; o > 0; o >>= 1) s += __shfl_xor(s, o, 32);
    return s;
}

__global__ __launch_bounds__(128)
void l2norm_rows(const float* __restrict__ x, float* y, int Nn, int F) {
    int r = blockIdx.x * 4 + (threadIdx.x >> 5);
    if (r >= Nn) return;
    int lane = threadIdx.x & 31;
    const float* xs = x + (size_t)r * F;
    float s = 0.f;
    for (int f = lane; f < F; f += 32) { float v = xs[f]; s += v * v; }
    s = wave_sum(s);
    float inv = rsqrtf(s + 1e-12f);
    float* ys = y + (size_t)r * F;
    for (int f = lane; f < F; f += 32) ys[f] = xs[f] * inv;
}

__global__ __launch_bounds__(128)
void edge_sim(const int* __restrict__ nrow, const int* __restrict__ ncol,
              const float* __restrict__ sn, float* sim, int E, int F) {
    int e = blockIdx.x * 4 + (threadIdx.x >> 5);
    if (e >= E) return;
    int lane = threadIdx.x & 31;
    const float* a = sn + (size_t)nrow[e] * F;
    const float* b = sn + (size_t)ncol[e] * F;
    float s = 0.f;
    for (int f = lane; f < F; f += 32) s += a[f] * b[f];
    s = wave_sum(s);
    if (lane == 0) sim[e] = s;
}

// ---------------------------------------------------------------- top-3 (chunked)
__global__ void s_scatter(const int* __restrict__ nrow, const int* __restrict__ ncol,
                          const float* __restrict__ sim, float* Schunk,
                          unsigned char* Achunk, int E, int c0, int ch, int Nn) {
    int e = blockIdx.x * blockDim.x + threadIdx.x;
    if (e >= E) return;
    int r = nrow[e];
    if (r < c0 || r >= c0 + ch) return;
    size_t off = (size_t)(r - c0) * Nn + ncol[e];
    atomicAdd(&Schunk[off], sim[e]);
    Achunk[off] = 1;
}

__device__ __forceinline__ bool better(float va, int ia, float vb, int ib) {
    return (va > vb) || (va == vb && (unsigned)ia < (unsigned)ib);
}

__global__ __launch_bounds__(256)
void top3_rows(const float* __restrict__ Schunk, const unsigned char* __restrict__ Achunk,
               int* knn, unsigned char* kmask, int c0, int Nn) {
    int rloc = blockIdx.x;
    int row  = c0 + rloc;
    const float* Srow = Schunk + (size_t)rloc * Nn;
    int tid = threadIdx.x;

    float v0 = NEG_HUGE, v1 = NEG_HUGE, v2 = NEG_HUGE;
    int   i0 = -1, i1 = -1, i2 = -1;
    for (int c = tid; c < Nn; c += 256) {
        float v = Srow[c];
        if (better(v, c, v0, i0)) { v2 = v1; i2 = i1; v1 = v0; i1 = i0; v0 = v; i0 = c; }
        else if (better(v, c, v1, i1)) { v2 = v1; i2 = i1; v1 = v; i1 = c; }
        else if (better(v, c, v2, i2)) { v2 = v; i2 = c; }
    }
    __shared__ float sv[256 * 3];
    __shared__ int   si[256 * 3];
    sv[tid * 3 + 0] = v0; si[tid * 3 + 0] = i0;
    sv[tid * 3 + 1] = v1; si[tid * 3 + 1] = i1;
    sv[tid * 3 + 2] = v2; si[tid * 3 + 2] = i2;
    for (int off = 128; off > 0; off >>= 1) {
        __syncthreads();
        if (tid < off) {
            float av[3], bv[3], mv[3]; int ai[3], bi[3], mi[3];
            #pragma unroll
            for (int k = 0; k < 3; ++k) {
                av[k] = sv[tid * 3 + k];          ai[k] = si[tid * 3 + k];
                bv[k] = sv[(tid + off) * 3 + k];  bi[k] = si[(tid + off) * 3 + k];
            }
            int a = 0, b = 0;
            #pragma unroll
            for (int k = 0; k < 3; ++k) {     // a+b==k<=2 so indices stay in range
                if (better(av[a], ai[a], bv[b], bi[b])) { mv[k] = av[a]; mi[k] = ai[a]; ++a; }
                else                                    { mv[k] = bv[b]; mi[k] = bi[b]; ++b; }
            }
            #pragma unroll
            for (int k = 0; k < 3; ++k) { sv[tid * 3 + k] = mv[k]; si[tid * 3 + k] = mi[k]; }
        }
    }
    __syncthreads();
    if (tid == 0) {
        #pragma unroll
        for (int k = 0; k < 3; ++k) {
            int nb = si[k];
            knn[row * 3 + k]   = nb;
            kmask[row * 3 + k] = (nb >= 0) ? Achunk[(size_t)rloc * Nn + nb] : 0;
        }
    }
}

// ---------------------------------------------------------------- loss terms
__global__ __launch_bounds__(128)
void term0_kernel(const float* __restrict__ pn, const float* __restrict__ tgt,
                  float* acc, int Nn, int F) {
    int r = blockIdx.x * 4 + (threadIdx.x >> 5);
    if (r >= Nn) return;
    int lane = threadIdx.x & 31;
    float s = 0.f;
    for (int f = lane; f < F; f += 32) s += pn[(size_t)r * F + f] * tgt[(size_t)r * F + f];
    s = wave_sum(s);
    if (lane == 0) atomicAdd(&acc[0], 2.f - 2.f * s);
}

__global__ __launch_bounds__(128)
void termk_kernel(const float* __restrict__ pn, const float* __restrict__ tgt,
                  const int* __restrict__ knn, const unsigned char* __restrict__ kmask,
                  float* acc, int which, int Nn, int F) {
    int r = blockIdx.x * 4 + (threadIdx.x >> 5);
    if (r >= Nn) return;
    int lane = threadIdx.x & 31;
    int nb = knn[r * 3 + which];
    float m = kmask[r * 3 + which] ? 1.f : 0.f;
    float s = 0.f;
    for (int f = lane; f < F; f += 32) s += pn[(size_t)r * F + f] * tgt[(size_t)nb * F + f];
    s = wave_sum(s);
    if (lane == 0) {
        atomicAdd(&acc[1 + which * 2], m * (2.f - 2.f * s));
        atomicAdd(&acc[2 + which * 2], m);
    }
}

__global__ void finalize_loss(const float* __restrict__ acc, float* out, int Nn) {
    float total = acc[0] / (float)Nn;
    #pragma unroll
    for (int i = 0; i < 3; ++i)
        total += acc[1 + i * 2] / fmaxf(acc[2 + i * 2], 1.f);
    out[0] = total / 4.f;
}

// ================================================================ launcher
extern "C" void kernel_launch(void* const* d_in, const int* in_sizes, int n_in,
                              void* d_out, int out_size, void* d_ws, size_t ws_size,
                              hipStream_t stream) {
    (void)n_in; (void)out_size; (void)ws_size;
    const float* x   = (const float*)d_in[0];
    const int*   edg = (const int*)d_in[1];
    const int*   nbr = (const int*)d_in[2];
    const float* W1  = (const float*)d_in[3];
    const float* b1  = (const float*)d_in[4];
    const float* g1  = (const float*)d_in[5];
    const float* bt1 = (const float*)d_in[6];
    const float* pr1 = (const float*)d_in[7];
    const float* W2  = (const float*)d_in[8];
    const float* b2  = (const float*)d_in[9];
    const float* g2  = (const float*)d_in[10];
    const float* bt2 = (const float*)d_in[11];
    const float* pr2 = (const float*)d_in[12];
    const float* pW1 = (const float*)d_in[13];
    const float* pb1 = (const float*)d_in[14];
    const float* pg  = (const float*)d_in[15];
    const float* pbb = (const float*)d_in[16];
    const float* ppr = (const float*)d_in[17];
    const float* pW2 = (const float*)d_in[18];
    const float* pb2 = (const float*)d_in[19];

    const int Nn = in_sizes[0] / D_IN;   // 10000
    const int E  = in_sizes[1] / 2;      // 320000
    const int* erow = edg;       const int* ecol = edg + E;
    const int* nrow = nbr;       const int* ncol = nbr + E;

    char* p = (char*)d_ws;
    auto carve = [&](size_t bytes) -> char* {
        char* r = p; p += (bytes + 255) & ~(size_t)255; return r;
    };
    float* deg   = (float*)carve((size_t)Nn * 4);
    float* dis   = (float*)carve((size_t)Nn * 4);
    float* bufA  = (float*)carve((size_t)Nn * 512 * 4);   // h / pact
    float* bufB  = (float*)carve((size_t)Nn * 512 * 4);   // agg / pred
    float* bufC  = (float*)carve((size_t)Nn * 512 * 4);   // act1 / ph
    float* sn    = (float*)carve((size_t)Nn * REP * 4);
    float* pn    = (float*)carve((size_t)Nn * REP * 4);
    float* mean  = (float*)carve(512 * 4);
    float* rstd  = (float*)carve(512 * 4);
    float* sim   = (float*)carve((size_t)E * 4);
    float* Schnk = (float*)carve((size_t)CHUNK * Nn * 4);
    __bf16* abf  = (__bf16*)carve((size_t)Nn * 512 * 2);  // bf16 activations
    __bf16* wtbf = (__bf16*)carve((size_t)512 * 512 * 2); // bf16 transposed weight
    unsigned char* Achnk = (unsigned char*)carve((size_t)CHUNK * Nn);
    int*   knn   = (int*)carve((size_t)Nn * 3 * 4);
    unsigned char* kmask = (unsigned char*)carve((size_t)Nn * 3);
    float* acc   = (float*)carve(8 * 4);

    float* student = (float*)d_out;
    float* lossOut = student + (size_t)Nn * REP;

    auto gemm = [&](const float* Af32, const float* W, float* C, const float* bias,
                    int M, int K, int Ncols) {
        size_t an = (size_t)M * K;
        cvt_f32_bf16<<<(int)((an + 255) / 256), 256, 0, stream>>>(Af32, abf, an);
        cvt_transpose_bf16<<<(K * Ncols + 255) / 256, 256, 0, stream>>>(W, wtbf, K, Ncols);
        int tiles  = (M / 16) * (Ncols / 32);
        int blocks = (tiles + 7) / 8;
        gemm_bf16_wmma<<<blocks, 256, 0, stream>>>(abf, wtbf, C, bias, M, K, Ncols);
    };

    // ---- symmetric normalization coefficients
    deg_init <<<(Nn + 255) / 256, 256, 0, stream>>>(deg, Nn);
    deg_count<<<(E + 255) / 256, 256, 0, stream>>>(ecol, deg, E);
    deg_rsqrt<<<(Nn + 255) / 256, 256, 0, stream>>>(deg, dis, Nn);

    // ---- encoder layer 1 (teacher == student, computed once)
    gemm(x, W1, bufA, nullptr, Nn, D_IN, H1d);
    hipMemsetAsync(bufB, 0, (size_t)Nn * H1d * 4, stream);
    gcn_scatter<<<E, 128, 0, stream>>>(erow, ecol, dis, bufA, bufB, E, H1d);
    gcn_self_bias<<<(int)(((size_t)Nn * H1d + 255) / 256), 256, 0, stream>>>(dis, bufA, b1, bufB, Nn, H1d);
    bn_stats<<<H1d, 256, 0, stream>>>(bufB, mean, rstd, Nn, H1d);
    bn_prelu<<<(int)(((size_t)Nn * H1d + 255) / 256), 256, 0, stream>>>(bufB, mean, rstd, g1, bt1, pr1, bufC, Nn, H1d);

    // ---- encoder layer 2
    gemm(bufC, W2, bufA, nullptr, Nn, H1d, REP);
    hipMemsetAsync(bufB, 0, (size_t)Nn * REP * 4, stream);
    gcn_scatter<<<E, 128, 0, stream>>>(erow, ecol, dis, bufA, bufB, E, REP);
    gcn_self_bias<<<(int)(((size_t)Nn * REP + 255) / 256), 256, 0, stream>>>(dis, bufA, b2, bufB, Nn, REP);
    bn_stats<<<REP, 256, 0, stream>>>(bufB, mean, rstd, Nn, REP);
    bn_prelu<<<(int)(((size_t)Nn * REP + 255) / 256), 256, 0, stream>>>(bufB, mean, rstd, g2, bt2, pr2, student, Nn, REP);

    // ---- predictor
    gemm(student, pW1, bufC, pb1, Nn, REP, PHID);
    bn_stats<<<PHID, 256, 0, stream>>>(bufC, mean, rstd, Nn, PHID);
    bn_prelu<<<(int)(((size_t)Nn * PHID + 255) / 256), 256, 0, stream>>>(bufC, mean, rstd, pg, pbb, ppr, bufA, Nn, PHID);
    gemm(bufA, pW2, bufB, pb2, Nn, PHID, REP);     // pred -> bufB

    // ---- normalized embeddings (tn == sn since teacher == student)
    l2norm_rows<<<(Nn + 3) / 4, 128, 0, stream>>>(student, sn, Nn, REP);
    l2norm_rows<<<(Nn + 3) / 4, 128, 0, stream>>>(bufB,    pn, Nn, REP);

    // ---- per-edge similarity
    edge_sim<<<(E + 3) / 4, 128, 0, stream>>>(nrow, ncol, sn, sim, E, REP);

    // ---- chunked dense top-3 of scattered S (stays in 5 MB scratch)
    for (int c0 = 0; c0 < Nn; c0 += CHUNK) {
        int ch = (Nn - c0 < CHUNK) ? (Nn - c0) : CHUNK;
        hipMemsetAsync(Schnk, 0, (size_t)ch * Nn * 4, stream);
        hipMemsetAsync(Achnk, 0, (size_t)ch * Nn, stream);
        s_scatter<<<(E + 255) / 256, 256, 0, stream>>>(nrow, ncol, sim, Schnk, Achnk, E, c0, ch, Nn);
        top3_rows<<<ch, 256, 0, stream>>>(Schnk, Achnk, knn, kmask, c0, Nn);
    }

    // ---- loss
    hipMemsetAsync(acc, 0, 8 * 4, stream);
    term0_kernel<<<(Nn + 3) / 4, 128, 0, stream>>>(pn, sn, acc, Nn, REP);
    for (int t = 0; t < 3; ++t)
        termk_kernel<<<(Nn + 3) / 4, 128, 0, stream>>>(pn, sn, knn, kmask, acc, t, Nn, REP);
    finalize_loss<<<1, 1, 0, stream>>>(acc, lossOut, Nn);
}
